// MLP_GCN_79439715107025
// MI455X (gfx1250) — compile-verified, matrix operand order checked by reading, not run on hardware
//
#include <hip/hip_runtime.h>

// ---------------------------------------------------------------------------
// MLP_GCN on gfx1250: f16 WMMA GEMMs with fused pre/post ops + atomic scatter
// ---------------------------------------------------------------------------

typedef __attribute__((ext_vector_type(16))) _Float16 v16h;
typedef __attribute__((ext_vector_type(8)))  float    v8f;

#define DIN     512
#define HH      64

// A-side pre-op (applied to f32 A elements before f16 conversion)
#define PRE_NONE       0
#define PRE_BIAS_SELU  1
// Epilogue modes
#define EP_SELFLOOP    0   // store raw C, and C2 = dinv[row]^2 * C (agg seed)
#define EP_BIAS_SELU   1   // store selu(C + bias)
#define EP_BIAS        2   // store C + bias

__device__ __forceinline__ float selu_f(float x) {
    const float alpha = 1.6732632423543772f;
    const float scale = 1.0507009873554805f;
    return scale * (x > 0.0f ? x : alpha * (expf(x) - 1.0f));
}

__device__ __forceinline__ float4 selu4(float4 v, float4 b) {
    float4 r;
    r.x = selu_f(v.x + b.x); r.y = selu_f(v.y + b.y);
    r.z = selu_f(v.z + b.z); r.w = selu_f(v.w + b.w);
    return r;
}

// ---------------------------------------------------------------------------
// Pack W[K x 64] (f32, row-major) into WMMA B-fragment order (f16).
//   lane = (n%16) + 16*g ; element e <-> kr = g*8+e (e<8), 16+g*8+(e-8) (e>=8)
// Stored as Bp[((ks*4+nt)*32 + lane)*16 + e]  -> 32B per lane, coalesced.
// ---------------------------------------------------------------------------
__global__ void pack_w_kernel(const float* __restrict__ W,
                              _Float16* __restrict__ Bp, int K) {
    int idx = blockIdx.x * 256 + threadIdx.x;      // over K*64
    if (idx >= K * 64) return;
    int k  = idx >> 6;
    int n  = idx & 63;
    int ks = k >> 5;
    int kr = k & 31;
    int g  = (kr >> 3) & 1;
    int e  = (kr & 7) + ((kr >> 4) << 3);
    int lane = (n & 15) + (g << 4);
    int nt   = n >> 4;
    Bp[((((ks << 2) + nt) << 5) + lane) * 16 + e] = (_Float16)W[idx];
}

// ---------------------------------------------------------------------------
// Degree / norm kernels
// ---------------------------------------------------------------------------
__global__ void deg_init_kernel(float* __restrict__ deg, int n) {
    int i = blockIdx.x * 256 + threadIdx.x;
    if (i < n) deg[i] = 1.0f;                       // self-loop weight
}

__global__ void deg_edges_kernel(const int* __restrict__ col,
                                 const float* __restrict__ w,
                                 float* __restrict__ deg, int e) {
    int i = blockIdx.x * 256 + threadIdx.x;
    if (i < e) atomicAdd(&deg[col[i]], w[i]);
}

__global__ void dinv_kernel(float* __restrict__ deg, int n) {
    int i = blockIdx.x * 256 + threadIdx.x;
    if (i < n) deg[i] = rsqrtf(deg[i]);             // deg >= 1 always
}

__global__ void norm_kernel(const int* __restrict__ row,
                            const int* __restrict__ col,
                            const float* __restrict__ w,
                            const float* __restrict__ dinv,
                            float* __restrict__ nrm, int e) {
    int i = blockIdx.x * 256 + threadIdx.x;
    if (i < e) nrm[i] = dinv[row[i]] * w[i] * dinv[col[i]];
}

// ---------------------------------------------------------------------------
// Edge scatter: agg[col][f] += norm * h[row][f]   ((edge,feature) parallel)
// ---------------------------------------------------------------------------
__global__ void agg_edges_kernel(const int* __restrict__ row,
                                 const int* __restrict__ col,
                                 const float* __restrict__ nrm,
                                 const float* __restrict__ h,
                                 float* __restrict__ agg, int e) {
    int idx = blockIdx.x * 256 + threadIdx.x;       // over e*64
    if (idx >= e * 64) return;
    int ei = idx >> 6;
    int f  = idx & 63;
    float v = nrm[ei] * h[row[ei] * 64 + f];
    atomicAdd(&agg[col[ei] * 64 + f], v);
}

// ---------------------------------------------------------------------------
// WMMA GEMM: C[M x 64] = f(A)[M x K] @ Bpack (f16 fragments)
// One wave computes a 16-row x 64-col strip: 1 A-fragment reused for 4 tiles.
// PRE: optional selu(x + preBias[k]) applied to A during f32->f16 conversion.
// EP : SELFLOOP writes raw C plus C2 = dinv[row]^2 * C; others fuse bias/selu.
// K multiple of 32; M multiple of 16 (50000 = 3125*16).
// ---------------------------------------------------------------------------
template <int PRE, int EP>
__global__ void gemm_wmma_kernel(const float* __restrict__ A,
                                 const _Float16* __restrict__ Bp,
                                 const float* __restrict__ preBias,
                                 const float* __restrict__ bias,
                                 const float* __restrict__ dinv,
                                 float* __restrict__ C,
                                 float* __restrict__ C2,
                                 int M, int K) {
    const int wave = blockIdx.x * (blockDim.x >> 5) + (threadIdx.x >> 5);
    const int lane = threadIdx.x & 31;
    const int m0   = wave << 4;
    if (m0 >= M) return;                            // wave-uniform guard

    const int g    = lane >> 4;                     // lane half
    const int mrow = m0 + (lane & 15);              // A row for this lane

    v8f acc0 = {}, acc1 = {}, acc2 = {}, acc3 = {};
    const int nks = K >> 5;

    for (int ks = 0; ks < nks; ++ks) {
        const int kk = ks << 5;
        const int c0 = kk + (g << 3);               // cols of f0/f1
        const int c1 = c0 + 16;                     // cols of f2/f3
        const float* ap = A + (size_t)mrow * K + c0;
        float4 f0 = *(const float4*)(ap);
        float4 f1 = *(const float4*)(ap + 4);
        float4 f2 = *(const float4*)(ap + 16);
        float4 f3 = *(const float4*)(ap + 20);
        if (kk + 32 < K) __builtin_prefetch(ap + 32, 0, 1);  // global_prefetch_b8

        if (PRE == PRE_BIAS_SELU) {
            f0 = selu4(f0, *(const float4*)(preBias + c0));
            f1 = selu4(f1, *(const float4*)(preBias + c0 + 4));
            f2 = selu4(f2, *(const float4*)(preBias + c1));
            f3 = selu4(f3, *(const float4*)(preBias + c1 + 4));
        }

        v16h a;
        a[0]  = (_Float16)f0.x; a[1]  = (_Float16)f0.y;
        a[2]  = (_Float16)f0.z; a[3]  = (_Float16)f0.w;
        a[4]  = (_Float16)f1.x; a[5]  = (_Float16)f1.y;
        a[6]  = (_Float16)f1.z; a[7]  = (_Float16)f1.w;
        a[8]  = (_Float16)f2.x; a[9]  = (_Float16)f2.y;
        a[10] = (_Float16)f2.z; a[11] = (_Float16)f2.w;
        a[12] = (_Float16)f3.x; a[13] = (_Float16)f3.y;
        a[14] = (_Float16)f3.z; a[15] = (_Float16)f3.w;

        const v16h* bptr = (const v16h*)(Bp + ((((size_t)ks << 2) << 5) + lane) * 16);
        v16h b0 = bptr[0];                          // tile stride = 32 v16h
        v16h b1 = bptr[32];
        v16h b2 = bptr[64];
        v16h b3 = bptr[96];

        acc0 = __builtin_amdgcn_wmma_f32_16x16x32_f16(false, a, false, b0, (short)0, acc0, false, false);
        acc1 = __builtin_amdgcn_wmma_f32_16x16x32_f16(false, a, false, b1, (short)0, acc1, false, false);
        acc2 = __builtin_amdgcn_wmma_f32_16x16x32_f16(false, a, false, b2, (short)0, acc2, false, false);
        acc3 = __builtin_amdgcn_wmma_f32_16x16x32_f16(false, a, false, b3, (short)0, acc3, false, false);
    }

    // C/D layout: VGPR v -> row m0 + 8*g + v, column = ntile*16 + (lane&15).
    const int crow0 = m0 + (g << 3);
    const int ccol  = lane & 15;
    #pragma unroll
    for (int v = 0; v < 8; ++v) {
        const int r = crow0 + v;
        float vals[4] = { acc0[v], acc1[v], acc2[v], acc3[v] };
        float dd = 0.0f;
        if (EP == EP_SELFLOOP) { float d = dinv[r]; dd = d * d; }
        float* crow  = C  + (size_t)r * 64;
        float* crow2 = (EP == EP_SELFLOOP) ? (C2 + (size_t)r * 64) : nullptr;
        #pragma unroll
        for (int n = 0; n < 4; ++n) {
            int c = (n << 4) + ccol;
            float x = vals[n];
            if (EP == EP_BIAS_SELU || EP == EP_BIAS) x += bias[c];
            if (EP == EP_BIAS_SELU) x = selu_f(x);
            crow[c] = x;
            if (EP == EP_SELFLOOP) crow2[c] = dd * x;
        }
    }
}

// ---------------------------------------------------------------------------
// Host-side orchestration
// ---------------------------------------------------------------------------
extern "C" void kernel_launch(void* const* d_in, const int* in_sizes, int n_in,
                              void* d_out, int out_size, void* d_ws, size_t ws_size,
                              hipStream_t stream) {
    const float* x    = (const float*)d_in[0];
    const int*   ei   = (const int*)d_in[1];
    const float* ew   = (const float*)d_in[2];
    const float* gW1  = (const float*)d_in[3];
    const float* gb1  = (const float*)d_in[4];
    const float* gW2  = (const float*)d_in[5];
    const float* gb2  = (const float*)d_in[6];
    const float* W0   = (const float*)d_in[7];
    const float* b0   = (const float*)d_in[8];
    const float* W1   = (const float*)d_in[9];
    const float* b1   = (const float*)d_in[10];
    const float* W2   = (const float*)d_in[11];
    const float* b2   = (const float*)d_in[12];
    const float* W3   = (const float*)d_in[13];
    const float* b3   = (const float*)d_in[14];

    const int N = in_sizes[0] / DIN;      // 50000
    const int E = in_sizes[2];            // 1600000
    const int* erow = ei;                 // edge_index[0] (source)
    const int* ecol = ei + E;             // edge_index[1] (target)

    // ---- workspace carve-up (256B aligned) ----
    char* ws = (char*)d_ws;
    size_t off = 0;
    auto carve = [&](size_t bytes) -> char* {
        char* p = ws + off;
        off = (off + bytes + 255) & ~(size_t)255;
        return p;
    };
    float*    dinv  = (float*)   carve((size_t)N * 4);
    float*    nrm   = (float*)   carve((size_t)E * 4);
    _Float16* W1p   = (_Float16*)carve((size_t)DIN * 64 * 2);
    _Float16* W2p   = (_Float16*)carve((size_t)HH  * 64 * 2);
    _Float16* W0mp  = (_Float16*)carve((size_t)HH  * 64 * 2);
    _Float16* W1mp  = (_Float16*)carve((size_t)HH  * 64 * 2);
    _Float16* W2mp  = (_Float16*)carve((size_t)HH  * 64 * 2);
    _Float16* W3mp  = (_Float16*)carve((size_t)HH  * 64 * 2);
    float*    bufA  = (float*)   carve((size_t)N * 64 * 4);
    float*    bufB  = (float*)   carve((size_t)N * 64 * 4);
    float*    bufC  = (float*)   carve((size_t)N * 64 * 4);
    (void)ws_size; (void)n_in;

    const int TPB = 256;
    auto blocks = [](long long n, int tpb) { return (int)((n + tpb - 1) / tpb); };

    // 1) pack weights into f16 WMMA fragment layout
    pack_w_kernel<<<blocks((long long)DIN * 64, TPB), TPB, 0, stream>>>(gW1, W1p, DIN);
    pack_w_kernel<<<blocks((long long)HH  * 64, TPB), TPB, 0, stream>>>(gW2, W2p, HH);
    pack_w_kernel<<<blocks((long long)HH  * 64, TPB), TPB, 0, stream>>>(W0,  W0mp, HH);
    pack_w_kernel<<<blocks((long long)HH  * 64, TPB), TPB, 0, stream>>>(W1,  W1mp, HH);
    pack_w_kernel<<<blocks((long long)HH  * 64, TPB), TPB, 0, stream>>>(W2,  W2mp, HH);
    pack_w_kernel<<<blocks((long long)HH  * 64, TPB), TPB, 0, stream>>>(W3,  W3mp, HH);

    // 2) degrees -> dinv -> per-edge norm
    deg_init_kernel <<<blocks(N, TPB), TPB, 0, stream>>>(dinv, N);
    deg_edges_kernel<<<blocks(E, TPB), TPB, 0, stream>>>(ecol, ew, dinv, E);
    dinv_kernel     <<<blocks(N, TPB), TPB, 0, stream>>>(dinv, N);
    norm_kernel     <<<blocks(E, TPB), TPB, 0, stream>>>(erow, ecol, ew, dinv, nrm, E);

    const int gemm_blocks = blocks((long long)(N / 16), 8);  // 8 waves per block
    const long long EF = (long long)E * 64;

    // 3) GCN layer 1: h1 = x@gW1 -> bufA, agg seed dinv^2*h1 -> bufB; scatter edges
    gemm_wmma_kernel<PRE_NONE, EP_SELFLOOP>
        <<<gemm_blocks, TPB, 0, stream>>>(x, W1p, nullptr, nullptr, dinv, bufA, bufB, N, DIN);
    agg_edges_kernel<<<blocks(EF, TPB), TPB, 0, stream>>>(erow, ecol, nrm, bufA, bufB, E);

    // 4) GCN layer 2: A = selu(bufB + gb1) fused; h2 -> bufC, agg seed -> bufA
    gemm_wmma_kernel<PRE_BIAS_SELU, EP_SELFLOOP>
        <<<gemm_blocks, TPB, 0, stream>>>(bufB, W2p, gb1, nullptr, dinv, bufC, bufA, N, HH);
    agg_edges_kernel<<<blocks(EF, TPB), TPB, 0, stream>>>(erow, ecol, nrm, bufC, bufA, E);

    // 5) MLP stack: first layer fuses selu(agg2 + gb2) on the A side
    gemm_wmma_kernel<PRE_BIAS_SELU, EP_BIAS_SELU>
        <<<gemm_blocks, TPB, 0, stream>>>(bufA, W0mp, gb2, b0, nullptr, bufB, nullptr, N, HH);
    gemm_wmma_kernel<PRE_NONE, EP_BIAS_SELU>
        <<<gemm_blocks, TPB, 0, stream>>>(bufB, W1mp, nullptr, b1, nullptr, bufC, nullptr, N, HH);
    gemm_wmma_kernel<PRE_NONE, EP_BIAS_SELU>
        <<<gemm_blocks, TPB, 0, stream>>>(bufC, W2mp, nullptr, b2, nullptr, bufB, nullptr, N, HH);
    gemm_wmma_kernel<PRE_NONE, EP_BIAS>
        <<<gemm_blocks, TPB, 0, stream>>>(bufB, W3mp, nullptr, b3, nullptr, (float*)d_out, nullptr, N, HH);
    (void)out_size;
}
